// AttentionBlock_32083405701227
// MI455X (gfx1250) — matrix operand order repaired and checked by reading, not compiled, not run on hardware
//
#include <hip/hip_runtime.h>
#include <hip/hip_bf16.h>

typedef __attribute__((ext_vector_type(16))) __bf16 v16bf;
typedef __attribute__((ext_vector_type(8)))  float  v8f;
typedef unsigned short ushort_t;
typedef unsigned int   uint_t;

#define C_DIM 512
#define T_DIM 1024
#define B_DIM 16
#define NHEAD 8
#define CH    64
#define QK_SCALE 0.3535533905932738f   // 1/sqrt(sqrt(64))

// ---------- helpers ----------

__device__ __forceinline__ ushort_t f2bf(float f) {
    uint_t u = __float_as_uint(f);
    u = (u + 0x7FFFu + ((u >> 16) & 1u)) >> 16;   // RNE
    return (ushort_t)u;
}

union FragU { uint4 u[2]; v16bf v; };

// A fragment, 16x32 bf16 (M x K): lane holds row M=lane%16.
// half = lane>>4. K slices: [k0 + half*8, +8) and [k0 + 16 + half*8, +8).
__device__ __forceinline__ v16bf make_afrag(const ushort_t* rowp, int k0, int half) {
    FragU f;
    f.u[0] = *(const uint4*)(rowp + k0 + half * 8);
    f.u[1] = *(const uint4*)(rowp + k0 + 16 + half * 8);
    return f.v;
}

// B fragment, 32x16 bf16 (K x N): lane holds column N=lane%16, which is a
// contiguous row of the source matrix. K slices: [k0 + half*8, +8) and [k0 + 16 + half*8, +8).
__device__ __forceinline__ v16bf make_bfrag(const ushort_t* rowp, int k0, int half) {
    FragU f;
    f.u[0] = *(const uint4*)(rowp + k0 + half * 8);
    f.u[1] = *(const uint4*)(rowp + k0 + 16 + half * 8);
    return f.v;
}

__device__ __forceinline__ v8f wmma_bf16(v16bf a, v16bf b, v8f c) {
    return __builtin_amdgcn_wmma_f32_16x16x32_bf16(false, a, false, b, (short)0, c, false, false);
}

// Async DMA: copy 16 bytes from global memory into LDS (ASYNCcnt-tracked).
__device__ __forceinline__ void async_copy16(uint_t lds_addr, const ushort_t* gsrc) {
    unsigned long long ga = (unsigned long long)(uintptr_t)gsrc;
    asm volatile("global_load_async_to_lds_b128 %0, %1, off"
                 :: "v"(lds_addr), "v"(ga) : "memory");
}

__device__ __forceinline__ void wait_async0() {
    asm volatile("s_wait_asynccnt 0x0" ::: "memory");
}

// ---------- kernel 1: layernorm over T per (b,c); write xn f32 (b,c,t) and xnT bf16 (b,t,c) ----------

__global__ void ln_kernel(const float* __restrict__ x, float* __restrict__ xn,
                          ushort_t* __restrict__ xnT) {
    int bid = blockIdx.x;             // b*512 + c
    int b = bid >> 9, c = bid & 511;
    int tid = threadIdx.x;            // 256 threads, 4 elems each
    const float* row = x + (size_t)bid * T_DIM;

    float4 v = ((const float4*)row)[tid];
    float s = v.x + v.y + v.z + v.w;
    float q = v.x * v.x + v.y * v.y + v.z * v.z + v.w * v.w;

    __shared__ float s1[256], s2[256];
    s1[tid] = s; s2[tid] = q;
    __syncthreads();
    for (int off = 128; off > 0; off >>= 1) {
        if (tid < off) { s1[tid] += s1[tid + off]; s2[tid] += s2[tid + off]; }
        __syncthreads();
    }
    float mean = s1[0] * (1.0f / T_DIM);
    float var  = s2[0] * (1.0f / T_DIM) - mean * mean;
    float inv  = rsqrtf(var + 1e-5f);

    float4 o;
    o.x = (v.x - mean) * inv; o.y = (v.y - mean) * inv;
    o.z = (v.z - mean) * inv; o.w = (v.w - mean) * inv;
    ((float4*)(xn + (size_t)bid * T_DIM))[tid] = o;

    int t0 = tid * 4;
    size_t base = ((size_t)b * T_DIM) * C_DIM + c;
    xnT[base + (size_t)(t0 + 0) * C_DIM] = f2bf(o.x);
    xnT[base + (size_t)(t0 + 1) * C_DIM] = f2bf(o.y);
    xnT[base + (size_t)(t0 + 2) * C_DIM] = f2bf(o.z);
    xnT[base + (size_t)(t0 + 3) * C_DIM] = f2bf(o.w);
}

// ---------- kernel 2: f32 -> bf16 convert (weights) ----------

__global__ void cvt_kernel(const float* __restrict__ in, ushort_t* __restrict__ out, int n) {
    int i = blockIdx.x * blockDim.x + threadIdx.x;
    if (i < n) out[i] = f2bf(in[i]);
}

// ---------- kernel 3: QKV GEMM. D[t,o] = sum_c xnT[b,t,c] * w_qkv[o,c] + b_qkv[o] ----------
// grid (T/128, 3C/64, B), block 256 (8 waves). Wave = 16 t-rows x 64 o-cols
// (4 col-tiles -> 32 acc VGPRs, spill-free). Weight tile (64 o x 32 k, shared
// by all 8 waves) is double-buffered in LDS, filled by one async
// global->LDS b128 per thread, overlapping the WMMAs of the previous k-step.
// Epilogue: q,k (scaled) -> qkvT[b,t,1536]; v -> vsep[bh,ch,t].

__global__ void qkv_gemm_kernel(const ushort_t* __restrict__ xnT,
                                const ushort_t* __restrict__ wq,
                                const float* __restrict__ bias,
                                ushort_t* __restrict__ qkvT,
                                ushort_t* __restrict__ vsep) {
    int tid = threadIdx.x, wave = tid >> 5, lane = tid & 31;
    int half = lane >> 4, ln = lane & 15;
    int b = blockIdx.z;
    int mBase = blockIdx.x * 128 + wave * 16;     // t
    int nBase = blockIdx.y * 64;                  // o

    __shared__ ushort_t bsm[2][64][32];           // 8 KB double-buffered B tile

    const ushort_t* arow = xnT + ((size_t)(b * T_DIM + mBase + ln)) * C_DIM;

    // this thread's slice of the tile copy: 8 elements (16 B) of one weight row
    int crow  = tid >> 2;                         // 0..63  (o - nBase)
    int cpart = (tid & 3) << 3;                   // element offset 0/8/16/24
    const ushort_t* gsrc0 = wq + (size_t)(nBase + crow) * C_DIM + cpart;
    uint_t ldst[2] = { (uint_t)(uintptr_t)&bsm[0][crow][cpart],
                       (uint_t)(uintptr_t)&bsm[1][crow][cpart] };

    // prefetch stage 0
    async_copy16(ldst[0], gsrc0);
    wait_async0();
    __syncthreads();

    v8f acc[4] = {};
    for (int s = 0; s < 16; ++s) {
        int k0 = s * 32;
        int buf = s & 1;
        if (s < 15)
            async_copy16(ldst[buf ^ 1], gsrc0 + k0 + 32);

        v16bf a = make_afrag(arow, k0, half);
        #pragma unroll
        for (int ct = 0; ct < 4; ++ct)
            acc[ct] = wmma_bf16(a, make_bfrag(&bsm[buf][ct * 16 + ln][0], 0, half), acc[ct]);

        __syncthreads();                  // everyone done reading bsm[buf]
        if (s < 15) { wait_async0(); __syncthreads(); }   // next tile resident
    }

    #pragma unroll
    for (int ct = 0; ct < 4; ++ct) {
        int o = nBase + ct * 16 + ln;
        float bs = bias[o];
        bool isqk = (o < 2 * C_DIM);
        float sc = isqk ? QK_SCALE : 1.0f;
        #pragma unroll
        for (int r = 0; r < 8; ++r) {
            int t = mBase + r + 8 * half;
            float val = (acc[ct][r] + bs) * sc;
            if (isqk) {
                qkvT[((size_t)(b * T_DIM + t)) * (3 * C_DIM) + o] = f2bf(val);
            } else {
                int chn = o - 2 * C_DIM;
                int hd = chn >> 6, cc = chn & 63;
                vsep[(((size_t)(b * NHEAD + hd)) * CH + cc) * T_DIM + t] = f2bf(val);
            }
        }
    }
}

// ---------- kernel 4: fused attention per (t-tile of 16, bh) ----------
// Stage 1: all 8 waves compute 16x1024 scores (each wave a 128-wide s chunk), K=64.
// Stage 2: cross-wave softmax stats (shfl_xor within 16-lane halves + LDS),
//          store unnormalized exp() as bf16 probs in LDS (never touches HBM).
// Stage 3: waves 0..3 compute aT[t, ch] = P @ V over K=1024 reading A-frags from LDS;
//          normalization (1/rowsum) folded into epilogue.

__global__ void attn_kernel(const ushort_t* __restrict__ qkvT,
                            const ushort_t* __restrict__ vsep,
                            ushort_t* __restrict__ aT) {
    int tid = threadIdx.x, wave = tid >> 5, lane = tid & 31;
    int half = lane >> 4, ln = lane & 15;
    int tBase = blockIdx.x * 16;
    int bh = blockIdx.y;
    int b = bh >> 3, h = bh & 7;

    __shared__ ushort_t pb[16 * 1024];       // unnormalized probs, bf16 (32 KB)
    __shared__ float    part[8][16];
    __shared__ float    rowmax[16];
    __shared__ float    rinv[16];

    // ---- stage 1: scores ----
    int sBase = wave * 128;
    const ushort_t* qrow =
        qkvT + ((size_t)(b * T_DIM + tBase + ln)) * (3 * C_DIM) + h * CH;

    v8f acc[8] = {};
    #pragma unroll
    for (int k0 = 0; k0 < CH; k0 += 32) {
        v16bf a = make_afrag(qrow, k0, half);
        #pragma unroll
        for (int ct = 0; ct < 8; ++ct) {
            const ushort_t* krow =
                qkvT + ((size_t)(b * T_DIM + sBase + ct * 16 + ln)) * (3 * C_DIM)
                     + C_DIM + h * CH;
            acc[ct] = wmma_bf16(a, make_bfrag(krow, k0, half), acc[ct]);
        }
    }

    // ---- stage 2a: row max (rows r+8*half per lane-half) ----
    float lmax[8];
    #pragma unroll
    for (int r = 0; r < 8; ++r) {
        float m = acc[0][r];
        #pragma unroll
        for (int ct = 1; ct < 8; ++ct) m = fmaxf(m, acc[ct][r]);
        lmax[r] = m;
    }
    #pragma unroll
    for (int mask = 1; mask < 16; mask <<= 1)
        #pragma unroll
        for (int r = 0; r < 8; ++r)
            lmax[r] = fmaxf(lmax[r], __shfl_xor(lmax[r], mask, 32));
    if (ln == 0)
        #pragma unroll
        for (int r = 0; r < 8; ++r) part[wave][r + 8 * half] = lmax[r];
    __syncthreads();
    if (tid < 16) {
        float m = part[0][tid];
        #pragma unroll
        for (int w = 1; w < 8; ++w) m = fmaxf(m, part[w][tid]);
        rowmax[tid] = m;
    }
    __syncthreads();

    // ---- stage 2b: exp, store bf16 to LDS, row sums ----
    float lsum[8] = {0.f, 0.f, 0.f, 0.f, 0.f, 0.f, 0.f, 0.f};
    #pragma unroll
    for (int ct = 0; ct < 8; ++ct) {
        int s = sBase + ct * 16 + ln;
        #pragma unroll
        for (int r = 0; r < 8; ++r) {
            int m = r + 8 * half;
            float e = __expf(acc[ct][r] - rowmax[m]);
            lsum[r] += e;
            pb[m * 1024 + s] = f2bf(e);
        }
    }
    #pragma unroll
    for (int mask = 1; mask < 16; mask <<= 1)
        #pragma unroll
        for (int r = 0; r < 8; ++r) lsum[r] += __shfl_xor(lsum[r], mask, 32);
    if (ln == 0)
        #pragma unroll
        for (int r = 0; r < 8; ++r) part[wave][r + 8 * half] = lsum[r];
    __syncthreads();
    if (tid < 16) {
        float s = 0.f;
        #pragma unroll
        for (int w = 0; w < 8; ++w) s += part[w][tid];
        rinv[tid] = 1.0f / s;
    }
    __syncthreads();

    // ---- stage 3: aT = P @ V (waves 0..3, each one 16-wide ch tile) ----
    if (wave < 4) {
        int ch = wave * 16 + ln;
        const ushort_t* vrow = vsep + (((size_t)bh) * CH + ch) * T_DIM;
        const ushort_t* prow = pb + ln * 1024;
        v8f acc2 = {};
        for (int k0 = 0; k0 < T_DIM; k0 += 32) {
            v16bf a = make_afrag(prow, k0, half);     // ds_load_b128 from LDS
            acc2 = wmma_bf16(a, make_bfrag(vrow, k0, half), acc2);
        }
        #pragma unroll
        for (int r = 0; r < 8; ++r) {
            int m = r + 8 * half;
            int t = tBase + m;
            float val = acc2[r] * rinv[m];
            aT[((size_t)(b * T_DIM + t)) * C_DIM + h * CH + ch] = f2bf(val);
        }
    }
}

// ---------- kernel 5: proj GEMM + residual (LDS-staged weight tile, async DMA) ----------
// out[b,o,t] = xn[b,o,t] + sum_c aT[b,t,c]*w_proj[o,c] + b_proj[o]

__global__ void proj_gemm_kernel(const ushort_t* __restrict__ aT,
                                 const ushort_t* __restrict__ wp,
                                 const float* __restrict__ bias,
                                 const float* __restrict__ xn,
                                 float* __restrict__ out) {
    int tid = threadIdx.x, wave = tid >> 5, lane = tid & 31;
    int half = lane >> 4, ln = lane & 15;
    int b = blockIdx.z;
    int mBase = blockIdx.x * 128 + wave * 16;     // t
    int nBase = blockIdx.y * 64;                  // o

    __shared__ ushort_t bsm[2][64][32];

    const ushort_t* arow = aT + ((size_t)(b * T_DIM + mBase + ln)) * C_DIM;

    int crow  = tid >> 2;
    int cpart = (tid & 3) << 3;
    const ushort_t* gsrc0 = wp + (size_t)(nBase + crow) * C_DIM + cpart;
    uint_t ldst[2] = { (uint_t)(uintptr_t)&bsm[0][crow][cpart],
                       (uint_t)(uintptr_t)&bsm[1][crow][cpart] };

    async_copy16(ldst[0], gsrc0);
    wait_async0();
    __syncthreads();

    v8f acc[4] = {};
    for (int s = 0; s < 16; ++s) {
        int k0 = s * 32;
        int buf = s & 1;
        if (s < 15)
            async_copy16(ldst[buf ^ 1], gsrc0 + k0 + 32);

        v16bf a = make_afrag(arow, k0, half);
        #pragma unroll
        for (int ct = 0; ct < 4; ++ct)
            acc[ct] = wmma_bf16(a, make_bfrag(&bsm[buf][ct * 16 + ln][0], 0, half), acc[ct]);

        __syncthreads();
        if (s < 15) { wait_async0(); __syncthreads(); }
    }

    int tb = mBase + 8 * half;                    // 8 consecutive t per lane
    #pragma unroll
    for (int ct = 0; ct < 4; ++ct) {
        int o = nBase + ct * 16 + ln;
        float bs = bias[o];
        size_t rowoff = ((size_t)(b * C_DIM + o)) * T_DIM + tb;
        float4 x0 = *(const float4*)(xn + rowoff);
        float4 x1 = *(const float4*)(xn + rowoff + 4);
        float4 o0, o1;
        o0.x = x0.x + acc[ct][0] + bs; o0.y = x0.y + acc[ct][1] + bs;
        o0.z = x0.z + acc[ct][2] + bs; o0.w = x0.w + acc[ct][3] + bs;
        o1.x = x1.x + acc[ct][4] + bs; o1.y = x1.y + acc[ct][5] + bs;
        o1.z = x1.z + acc[ct][6] + bs; o1.w = x1.w + acc[ct][7] + bs;
        *(float4*)(out + rowoff)     = o0;
        *(float4*)(out + rowoff + 4) = o1;
    }
}

// ---------- launch ----------

extern "C" void kernel_launch(void* const* d_in, const int* in_sizes, int n_in,
                              void* d_out, int out_size, void* d_ws, size_t ws_size,
                              hipStream_t stream) {
    const float* x      = (const float*)d_in[0];
    const float* w_qkv  = (const float*)d_in[1];
    const float* b_qkv  = (const float*)d_in[2];
    const float* w_proj = (const float*)d_in[3];
    const float* b_proj = (const float*)d_in[4];
    float* out = (float*)d_out;

    char* w = (char*)d_ws;
    float*    xn_f32 = (float*)w;     w += (size_t)B_DIM * C_DIM * T_DIM * 4;
    ushort_t* xnT    = (ushort_t*)w;  w += (size_t)B_DIM * T_DIM * C_DIM * 2;
    ushort_t* wq     = (ushort_t*)w;  w += (size_t)3 * C_DIM * C_DIM * 2;
    ushort_t* wp     = (ushort_t*)w;  w += (size_t)C_DIM * C_DIM * 2;
    ushort_t* qkvT   = (ushort_t*)w;  w += (size_t)B_DIM * T_DIM * 3 * C_DIM * 2;
    ushort_t* vsep   = (ushort_t*)w;  w += (size_t)B_DIM * NHEAD * CH * T_DIM * 2;
    ushort_t* aTbuf  = (ushort_t*)w;  w += (size_t)B_DIM * T_DIM * C_DIM * 2;

    ln_kernel<<<B_DIM * C_DIM, 256, 0, stream>>>(x, xn_f32, xnT);

    int nq = 3 * C_DIM * C_DIM;
    cvt_kernel<<<(nq + 255) / 256, 256, 0, stream>>>(w_qkv, wq, nq);
    int np = C_DIM * C_DIM;
    cvt_kernel<<<(np + 255) / 256, 256, 0, stream>>>(w_proj, wp, np);

    qkv_gemm_kernel<<<dim3(T_DIM / 128, (3 * C_DIM) / 64, B_DIM), 256, 0, stream>>>(
        xnT, wq, b_qkv, qkvT, vsep);

    attn_kernel<<<dim3(T_DIM / 16, B_DIM * NHEAD), 256, 0, stream>>>(qkvT, vsep, aTbuf);

    proj_gemm_kernel<<<dim3(T_DIM / 128, C_DIM / 64, B_DIM), 256, 0, stream>>>(
        aTbuf, wp, b_proj, xn_f32, out);
}
